// LinearAttention_46608985096564
// MI455X (gfx1250) — compile-verified, matrix operand order checked by reading, not run on hardware
//
#include <hip/hip_runtime.h>
#include <hip/hip_bf16.h>

// ---------------- types ----------------
typedef __bf16 bf16_t;
typedef __attribute__((ext_vector_type(16))) __bf16 v16bf;
typedef __attribute__((ext_vector_type(8)))  __bf16 v8bf;
typedef __attribute__((ext_vector_type(8)))  float  v8f;

union V8F  { v8f  v; float  f[8];  };
union V8B  { v8bf v; bf16_t h[8];  };
union V16B { v16bf v; bf16_t h[16]; };

#define DIMC  256
#define NPIX  4096
#define O3    384
#define HID   128
#define NB    32
#define QSCALE 0.17677669529663687f   /* 32^-0.5 */
#define EPSN  1e-5f

static __device__ inline v8f wmma_bf16(v16bf a, v16bf b, v8f c) {
  // D = A(16x32 bf16) * B(32x16 bf16) + C(16x16 f32)
  return __builtin_amdgcn_wmma_f32_16x16x32_bf16(false, a, false, b, (short)0, c,
                                                 false, false);
}
static __device__ inline v16bf join16(v8bf lo, v8bf hi) {
  return __builtin_shufflevector(lo, hi, 0,1,2,3,4,5,6,7,8,9,10,11,12,13,14,15);
}

// ---------------------------------------------------------------------------
// K0: pack w_qkv (384x256 fp32) into bf16 A-fragment order; zero GN stats.
// Layout: apack[Mt(24)][kc(8)][lane(32)][j(16)], lane<16 -> Kgroup 0, else +8.
// ---------------------------------------------------------------------------
__global__ void k_pack(const float* __restrict__ wqkv, bf16_t* __restrict__ apack,
                       float* __restrict__ gstat) {
  int e    = blockIdx.x * 256 + threadIdx.x;            // 0..98303
  int j    = e & 15;
  int lane = (e >> 4) & 31;
  int kc   = (e >> 9) & 7;
  int Mt   = e >> 12;
  int m    = Mt * 16 + (lane & 15);
  int g    = (lane < 16) ? 0 : 8;
  int k    = kc * 32 + ((j < 8) ? (g + j) : (16 + g + (j - 8)));
  apack[e] = (bf16_t)wqkv[m * DIMC + k];
  if (blockIdx.x == 0 && threadIdx.x < 64) gstat[threadIdx.x] = 0.0f;
}

// ---------------------------------------------------------------------------
// K1: QKV GEMM per (batch, 64-col n-tile): [384x256]x[256x64].
//     8 waves = 4(M-strips of 96) x 2(N-strips of 32); 12 acc tiles/wave.
//     Ping-pong LDS x-tile staging: next chunk's global loads issue before
//     the current chunk's WMMAs; cvt+ds_store_b128 happen after the WMMAs so
//     global-load latency overlaps matrix math.
//     Epilogue: q rows -> LDS, softmax over d, store q transposed (bf16);
//               k/v rows -> row-major bf16.
// ---------------------------------------------------------------------------
__global__ __launch_bounds__(256) void k_qkv(
    const float* __restrict__ x, const bf16_t* __restrict__ apack,
    bf16_t* __restrict__ qt, bf16_t* __restrict__ kf, bf16_t* __restrict__ vf) {
  __shared__ bf16_t Xs[2][64][32];  // [buf][n][c] transposed chunk, 8 KB
  __shared__ float  Ys[128][64];    // q rows restage, 32 KB

  int b     = blockIdx.x >> 6;
  int nbase = (blockIdx.x & 63) * 64;
  int tid   = threadIdx.x;
  int lane  = tid & 31;
  int wave  = tid >> 5;
  int wm    = wave >> 1;            // 0..3
  int wn    = wave & 1;             // 0..1
  int g     = (lane < 16) ? 0 : 8;
  int l15   = lane & 15;

  // staging mapping: thread = fixed n, 8 consecutive c -> one ds_store_b128
  int sc_n  = tid & 63;
  int sc_cb = (tid >> 6) * 8;

  v8f z8 = {0.f,0.f,0.f,0.f,0.f,0.f,0.f,0.f};
  v8f acc[6][2];
  #pragma unroll
  for (int mt = 0; mt < 6; ++mt) { acc[mt][0] = z8; acc[mt][1] = z8; }

  const float* xb = x + (size_t)b * DIMC * NPIX + nbase;

  // prologue: stage chunk 0 into buffer 0
  {
    const float* src = xb + (size_t)sc_cb * NPIX + sc_n;
    float xr[8];
    #pragma unroll
    for (int j = 0; j < 8; ++j) xr[j] = src[(size_t)j * NPIX];
    V8B u;
    #pragma unroll
    for (int j = 0; j < 8; ++j) u.h[j] = (bf16_t)xr[j];
    *(v8bf*)&Xs[0][sc_n][sc_cb] = u.v;
  }

  for (int kc = 0; kc < 8; ++kc) {
    __syncthreads();

    // issue next chunk's global loads into registers (no wait yet)
    float xn[8];
    if (kc < 7) {
      const float* src = xb + (size_t)((kc + 1) * 32 + sc_cb) * NPIX + sc_n;
      #pragma unroll
      for (int j = 0; j < 8; ++j) xn[j] = src[(size_t)j * NPIX];
    }

    int buf = kc & 1;
    v16bf bfr[2];
    #pragma unroll
    for (int nt = 0; nt < 2; ++nt) {
      int nl = wn * 32 + nt * 16 + l15;
      v8bf b0 = *(const v8bf*)&Xs[buf][nl][g];
      v8bf b1 = *(const v8bf*)&Xs[buf][nl][g + 16];
      bfr[nt] = join16(b0, b1);
    }
    v16bf afr[6];
    #pragma unroll
    for (int mt = 0; mt < 6; ++mt) {
      const bf16_t* ap = apack + ((size_t)((wm * 6 + mt) * 8 + kc) * 32 + lane) * 16;
      afr[mt] = join16(*(const v8bf*)ap, *(const v8bf*)(ap + 8));
    }
    #pragma unroll
    for (int mt = 0; mt < 6; ++mt) {
      acc[mt][0] = wmma_bf16(afr[mt], bfr[0], acc[mt][0]);
      acc[mt][1] = wmma_bf16(afr[mt], bfr[1], acc[mt][1]);
    }

    // convert + store next chunk into the other buffer (after the WMMAs)
    if (kc < 7) {
      V8B u;
      #pragma unroll
      for (int j = 0; j < 8; ++j) u.h[j] = (bf16_t)xn[j];
      *(v8bf*)&Xs[(kc + 1) & 1][sc_n][sc_cb] = u.v;
    }
  }

  // epilogue: q rows (m<128) to LDS; k (128..255), v (256..383) to global
  #pragma unroll
  for (int mt = 0; mt < 6; ++mt) {
    int Mt    = wm * 6 + mt;
    int mbase = Mt * 16 + g;
    #pragma unroll
    for (int nt = 0; nt < 2; ++nt) {
      int ncol = wn * 32 + nt * 16 + l15;
      V8F u; u.v = acc[mt][nt];
      if (mbase < 128) {
        #pragma unroll
        for (int r = 0; r < 8; ++r) Ys[mbase + r][ncol] = u.f[r];
      } else if (mbase < 256) {
        #pragma unroll
        for (int r = 0; r < 8; ++r)
          kf[((size_t)b * HID + (mbase + r - 128)) * NPIX + nbase + ncol] =
              (bf16_t)u.f[r];
      } else {
        #pragma unroll
        for (int r = 0; r < 8; ++r)
          vf[((size_t)b * HID + (mbase + r - 256)) * NPIX + nbase + ncol] =
              (bf16_t)u.f[r];
      }
    }
  }
  __syncthreads();

  { // q softmax over head-dim (32 rows) per column, *scale
    int h  = tid >> 6;              // 0..3
    int nl = tid & 63;
    float tv[32];
    float mx = -3.0e38f;
    #pragma unroll
    for (int d = 0; d < 32; ++d) { tv[d] = Ys[h * 32 + d][nl]; mx = fmaxf(mx, tv[d]); }
    float s = 0.0f;
    #pragma unroll
    for (int d = 0; d < 32; ++d) { tv[d] = __expf(tv[d] - mx); s += tv[d]; }
    float inv = QSCALE / s;
    #pragma unroll
    for (int d = 0; d < 32; ++d) Ys[h * 32 + d][nl] = tv[d] * inv;
  }
  __syncthreads();

  { // store q transposed: qt[b][n][o] bf16
    int nl = tid >> 2;
    int ob = (tid & 3) * 32;
    bf16_t* dst = qt + ((size_t)b * NPIX + nbase + nl) * HID + ob;
    #pragma unroll
    for (int i = 0; i < 32; i += 8) {
      V8B u;
      #pragma unroll
      for (int jj = 0; jj < 8; ++jj) u.h[jj] = (bf16_t)Ys[ob + i + jj][nl];
      *(v8bf*)(dst + i) = u.v;
    }
  }
}

// ---------------------------------------------------------------------------
// K2: per-row (b*128+o) max and 1/sum(exp) of k over n=4096. 1 block / row.
// ---------------------------------------------------------------------------
__global__ __launch_bounds__(256) void k_kstats(const bf16_t* __restrict__ kf,
                                                float* __restrict__ kmax,
                                                float* __restrict__ krcp) {
  __shared__ float red[256];
  int row = blockIdx.x;
  int tid = threadIdx.x;
  const bf16_t* src = kf + (size_t)row * NPIX + tid * 16;
  V8B u0, u1;
  u0.v = *(const v8bf*)src;
  u1.v = *(const v8bf*)(src + 8);
  float v[16];
  #pragma unroll
  for (int i = 0; i < 8; ++i) { v[i] = (float)u0.h[i]; v[8 + i] = (float)u1.h[i]; }
  float mx = v[0];
  #pragma unroll
  for (int i = 1; i < 16; ++i) mx = fmaxf(mx, v[i]);
  red[tid] = mx; __syncthreads();
  for (int s = 128; s > 0; s >>= 1) {
    if (tid < s) red[tid] = fmaxf(red[tid], red[tid + s]);
    __syncthreads();
  }
  mx = red[0]; __syncthreads();
  float sum = 0.0f;
  #pragma unroll
  for (int i = 0; i < 16; ++i) sum += __expf(v[i] - mx);
  red[tid] = sum; __syncthreads();
  for (int s = 128; s > 0; s >>= 1) {
    if (tid < s) red[tid] += red[tid + s];
    __syncthreads();
  }
  if (tid == 0) { kmax[row] = mx; krcp[row] = 1.0f / red[0]; }
}

// ---------------------------------------------------------------------------
// K3: raw context[b,h,d,e] = sum_n exp(k[d,n]-max_d) * v[e,n]  (1/Z in k_wc).
//     1 block per (b,h); 8 waves = 2(n-halves) x 2x2 tiles; halves combined
//     through LDS. Fragment loads software-pipelined one iteration ahead.
// ---------------------------------------------------------------------------
__global__ __launch_bounds__(256) void k_context(
    const bf16_t* __restrict__ kf, const bf16_t* __restrict__ vf,
    const float* __restrict__ kmax, float* __restrict__ ctx) {
  __shared__ __align__(32) float Ps[4][32][8];   // half=1 partials, 4 KB
  int bh   = blockIdx.x;            // b*4+h, rows of kf/vf are bh*32+d
  int tid  = threadIdx.x;
  int lane = tid & 31;
  int wave = tid >> 5;
  int half = wave >> 2;             // 0..1 : n range split
  int tile = wave & 3;
  int mi   = tile >> 1, ni = tile & 1;
  int g    = (lane < 16) ? 0 : 8;
  int l15  = lane & 15;

  const bf16_t* krow =
      kf + ((size_t)bh * 32 + mi * 16 + l15) * NPIX + half * 2048;
  const bf16_t* vrow =
      vf + ((size_t)bh * 32 + ni * 16 + l15) * NPIX + half * 2048;
  float km = kmax[bh * 32 + mi * 16 + l15];

  v8f acc = {0.f,0.f,0.f,0.f,0.f,0.f,0.f,0.f};

  // pipeline: prefetch iteration nc+32 before doing the exp work of nc
  v8bf pa0 = *(const v8bf*)(krow + g);
  v8bf pa1 = *(const v8bf*)(krow + g + 16);
  v8bf pb0 = *(const v8bf*)(vrow + g);
  v8bf pb1 = *(const v8bf*)(vrow + g + 16);
  for (int nc = 0; nc < 2048; nc += 32) {
    v8bf ca0 = pa0, ca1 = pa1, cb0 = pb0, cb1 = pb1;
    int nn = nc + 32;
    if (nn < 2048) {
      pa0 = *(const v8bf*)(krow + nn + g);
      pa1 = *(const v8bf*)(krow + nn + g + 16);
      pb0 = *(const v8bf*)(vrow + nn + g);
      pb1 = *(const v8bf*)(vrow + nn + g + 16);
    }
    V8B a0, a1; a0.v = ca0; a1.v = ca1;
    V16B af;
    #pragma unroll
    for (int i = 0; i < 8; ++i) {
      af.h[i]     = (bf16_t)__expf((float)a0.h[i] - km);
      af.h[8 + i] = (bf16_t)__expf((float)a1.h[i] - km);
    }
    acc = wmma_bf16(af.v, join16(cb0, cb1), acc);
  }

  if (half == 1) {
    *(v8f*)&Ps[tile][lane][0] = acc;
  }
  __syncthreads();
  if (half == 0) {
    V8F u; u.v = acc;
    V8F p; p.v = *(const v8f*)&Ps[tile][lane][0];
    int mb = mi * 16 + g;
    int e  = ni * 16 + l15;
    #pragma unroll
    for (int r = 0; r < 8; ++r)
      ctx[((size_t)bh * 32 + mb + r) * 32 + e] = u.f[r] + p.f[r];
  }
}

// ---------------------------------------------------------------------------
// K4a: Wc[b][c][h*32+d] = krcp[bhd] * sum_e w_out[c][h*32+e] * ctx[b,h,d,e].
// ---------------------------------------------------------------------------
__global__ void k_wc(const float* __restrict__ wout, const float* __restrict__ ctx,
                     const float* __restrict__ krcp, bf16_t* __restrict__ wc) {
  int t  = blockIdx.x * 256 + threadIdx.x;   // 32*256*128 total
  int hd = t & 127;
  int c  = (t >> 7) & 255;
  int b  = t >> 15;
  int h  = hd >> 5, d = hd & 31;
  const float* wr = wout + c * HID + h * 32;
  const float* cr = ctx + ((size_t)(b * 4 + h) * 32 + d) * 32;
  float s = 0.0f;
  #pragma unroll
  for (int e = 0; e < 32; ++e) s += wr[e] * cr[e];
  s *= krcp[(b * 4 + h) * 32 + d];
  wc[((size_t)b * DIMC + c) * HID + hd] = (bf16_t)s;
}

// ---------------------------------------------------------------------------
// K4b: y = Wc_b(256x128) @ q_b(128x4096) + b_out; also GN sum/sumsq atomics.
//      Both fragments direct from global (Wc row-major; q stored transposed);
//      per-k-chunk fragment loads hoisted so they clause up ahead of WMMAs.
// ---------------------------------------------------------------------------
__global__ __launch_bounds__(256) void k_y(
    const bf16_t* __restrict__ wc, const bf16_t* __restrict__ qt,
    const float* __restrict__ bout, float* __restrict__ y,
    float* __restrict__ gstat) {
  __shared__ float red[256];
  __shared__ float red2[256];
  int b     = blockIdx.x >> 6;
  int nbase = (blockIdx.x & 63) * 64;
  int tid   = threadIdx.x;
  int lane  = tid & 31;
  int wave  = tid >> 5;
  int wm    = wave >> 1, wn = wave & 1;
  int g     = (lane < 16) ? 0 : 8;
  int l15   = lane & 15;

  v8f z8 = {0.f,0.f,0.f,0.f,0.f,0.f,0.f,0.f};
  v8f acc[4][2];
  #pragma unroll
  for (int mt = 0; mt < 4; ++mt) { acc[mt][0] = z8; acc[mt][1] = z8; }

  #pragma unroll
  for (int kc = 0; kc < 4; ++kc) {
    v16bf bfr[2], afr[4];
    #pragma unroll
    for (int nt = 0; nt < 2; ++nt) {
      int n = nbase + wn * 32 + nt * 16 + l15;
      const bf16_t* qp = qt + ((size_t)b * NPIX + n) * HID + kc * 32 + g;
      bfr[nt] = join16(*(const v8bf*)qp, *(const v8bf*)(qp + 16));
    }
    #pragma unroll
    for (int mt = 0; mt < 4; ++mt) {
      int c = (wm * 4 + mt) * 16 + l15;
      const bf16_t* ap = wc + ((size_t)b * DIMC + c) * HID + kc * 32 + g;
      afr[mt] = join16(*(const v8bf*)ap, *(const v8bf*)(ap + 16));
    }
    #pragma unroll
    for (int mt = 0; mt < 4; ++mt) {
      acc[mt][0] = wmma_bf16(afr[mt], bfr[0], acc[mt][0]);
      acc[mt][1] = wmma_bf16(afr[mt], bfr[1], acc[mt][1]);
    }
  }

  float ls = 0.0f, ls2 = 0.0f;
  #pragma unroll
  for (int mt = 0; mt < 4; ++mt) {
    int cb = (wm * 4 + mt) * 16 + g;
    #pragma unroll
    for (int nt = 0; nt < 2; ++nt) {
      int n = nbase + wn * 32 + nt * 16 + l15;
      V8F u; u.v = acc[mt][nt];
      #pragma unroll
      for (int r = 0; r < 8; ++r) {
        int c = cb + r;
        float val = u.f[r] + bout[c];
        y[((size_t)b * DIMC + c) * NPIX + n] = val;
        ls += val; ls2 += val * val;
      }
    }
  }
  red[tid] = ls; red2[tid] = ls2;
  __syncthreads();
  for (int s = 128; s > 0; s >>= 1) {
    if (tid < s) { red[tid] += red[tid + s]; red2[tid] += red2[tid + s]; }
    __syncthreads();
  }
  if (tid == 0) {
    atomicAdd(&gstat[b], red[0]);
    atomicAdd(&gstat[32 + b], red2[0]);
  }
}

// ---------------------------------------------------------------------------
// K5: in-place GroupNorm(1) finalize on d_out.
// ---------------------------------------------------------------------------
__global__ void k_gn(float* __restrict__ y, const float* __restrict__ gstat,
                     const float* __restrict__ gw, const float* __restrict__ gb) {
  size_t e = ((size_t)blockIdx.x * 256 + threadIdx.x) * 4;
  int b = (int)(e >> 20);                 // 256*4096 = 2^20 per batch
  int c = (int)((e >> 12) & 255);
  const float invN = 1.0f / (float)(DIMC * NPIX);
  float mean = gstat[b] * invN;
  float var  = gstat[32 + b] * invN - mean * mean;
  float inv  = rsqrtf(var + EPSN);
  float sw = gw[c] * inv, sb = gb[c];
  float4* p = (float4*)(y + e);
  float4 v = *p;
  v.x = (v.x - mean) * sw + sb;
  v.y = (v.y - mean) * sw + sb;
  v.z = (v.z - mean) * sw + sb;
  v.w = (v.w - mean) * sw + sb;
  *p = v;
}

// ---------------------------------------------------------------------------
extern "C" void kernel_launch(void* const* d_in, const int* in_sizes, int n_in,
                              void* d_out, int out_size, void* d_ws, size_t ws_size,
                              hipStream_t stream) {
  (void)in_sizes; (void)n_in; (void)out_size; (void)ws_size;
  const float* x    = (const float*)d_in[0];
  const float* wqkv = (const float*)d_in[1];
  const float* wout = (const float*)d_in[2];
  const float* bout = (const float*)d_in[3];
  const float* gw   = (const float*)d_in[4];
  const float* gb   = (const float*)d_in[5];
  float* y = (float*)d_out;

  char* ws = (char*)d_ws;
  size_t off = 0;
  auto carve = [&](size_t bytes) {
    void* p = ws + off;
    off += (bytes + 255) & ~(size_t)255;
    return p;
  };
  bf16_t* apack = (bf16_t*)carve((size_t)O3 * DIMC * 2);        // 192 KB
  bf16_t* qt    = (bf16_t*)carve((size_t)NB * NPIX * HID * 2);  // 32 MB (transposed)
  bf16_t* kf    = (bf16_t*)carve((size_t)NB * HID * NPIX * 2);  // 32 MB
  bf16_t* vf    = (bf16_t*)carve((size_t)NB * HID * NPIX * 2);  // 32 MB
  float*  kmax  = (float*) carve((size_t)NB * HID * 4);
  float*  krcp  = (float*) carve((size_t)NB * HID * 4);
  float*  ctx   = (float*) carve((size_t)NB * 4 * 32 * 32 * 4); // 512 KB
  bf16_t* wcb   = (bf16_t*)carve((size_t)NB * DIMC * HID * 2);  // 2 MB
  float*  gstat = (float*) carve(64 * 4);

  k_pack   <<<384,    256, 0, stream>>>(wqkv, apack, gstat);
  k_qkv    <<<NB*64,  256, 0, stream>>>(x, apack, qt, kf, vf);
  k_kstats <<<NB*HID, 256, 0, stream>>>(kf, kmax, krcp);
  k_context<<<NB*4,   256, 0, stream>>>(kf, vf, kmax, ctx);
  k_wc     <<<4096,   256, 0, stream>>>(wout, ctx, krcp, wcb);
  k_y      <<<NB*64,  256, 0, stream>>>(wcb, qt, bout, y, gstat);
  k_gn     <<<32768,  256, 0, stream>>>(y, gstat, gw, gb);
}
